// DeepSeekMoELayer_76141180223864
// MI455X (gfx1250) — matrix operand order repaired and checked by reading, not compiled, not run on hardware
//
#include <hip/hip_runtime.h>

// Problem constants (match reference)
#define N_TOK 8192
#define HID   2048
#define DOUT  2048
#define NEXP  8

// GEMM tile config: 128x128x32, 256 threads (8 wave32), wave tile 32x64.
// Triple-buffered LDS filled by global_load_async_to_lds_b128, issue-ahead=2.
#define BM 128
#define BN 128
#define BK 32
#define LDSP 40   // padded row stride in bf16 elems (80B): 16B-aligned, spreads banks

typedef __attribute__((ext_vector_type(8)))  float  v8f;
typedef __attribute__((ext_vector_type(8)))  __bf16 v8bf;
typedef __attribute__((ext_vector_type(16))) __bf16 v16bf;

static __device__ __forceinline__ unsigned short f2bf(float f) {
    unsigned u = __builtin_bit_cast(unsigned, f);
    u += 0x7FFFu + ((u >> 16) & 1u);          // RNE fp32 -> bf16
    return (unsigned short)(u >> 16);
}
static __device__ __forceinline__ unsigned pack2(float a, float b) {
    return (unsigned)f2bf(a) | ((unsigned)f2bf(b) << 16);
}
// generic LDS pointer -> 32-bit LDS byte offset (aperture keeps it in addr[31:0])
static __device__ __forceinline__ unsigned lds_off(const void* p) {
    return (unsigned)(unsigned long long)p;
}
// async DMA 32B/lane (2 x b128) global -> LDS; offset applies to both sides
static __device__ __forceinline__ void async_copy_32B(unsigned ldsOff, unsigned vOff,
                                                      const unsigned short* sBase) {
    asm volatile("global_load_async_to_lds_b128 %0, %1, %2\n\t"
                 "global_load_async_to_lds_b128 %0, %1, %2 offset:16"
                 :: "v"(ldsOff), "v"(vOff), "s"(sBase) : "memory");
}
// wait until my wave's ASYNCcnt <= n (loads complete in order)
static __device__ __forceinline__ void wait_async0() {
    asm volatile("s_wait_asynccnt 0x0" ::: "memory");
}
static __device__ __forceinline__ void wait_async4() {
    asm volatile("s_wait_asynccnt 0x4" ::: "memory");
}

// ---------------------------------------------------------------------------
// One-time conversion: x fp32 [N,H] -> bf16 [N,H]. 8 elems/thread.
// ---------------------------------------------------------------------------
__global__ __launch_bounds__(256)
void convert_x_bf16(const float* __restrict__ src, unsigned short* __restrict__ dst)
{
    size_t i = ((size_t)blockIdx.x * 256 + threadIdx.x) * 8;
    float4 a = *(const float4*)(src + i);
    float4 b = *(const float4*)(src + i + 4);
    uint4 o;
    o.x = pack2(a.x, a.y); o.y = pack2(a.z, a.w);
    o.z = pack2(b.x, b.y); o.w = pack2(b.z, b.w);
    *(uint4*)(dst + i) = o;
}

// ---------------------------------------------------------------------------
// One-time conversion + transpose: W fp32 [H,D] -> bf16 [D,H].
// z==0: W_shared; z>=1: W_routed expert z-1. 32x32 tiles via LDS.
// ---------------------------------------------------------------------------
__global__ __launch_bounds__(256)
void convert_w_t_bf16(const float* __restrict__ Wsh, const float* __restrict__ Wrt,
                      unsigned short* __restrict__ WshT, unsigned short* __restrict__ WrtT)
{
    __shared__ __attribute__((aligned(8))) unsigned short tile[32][40];
    const int z = blockIdx.z;
    const float* src = (z == 0) ? Wsh : (Wrt + (size_t)(z - 1) * HID * DOUT);
    unsigned short* dst = (z == 0) ? WshT : (WrtT + (size_t)(z - 1) * HID * DOUT);
    const int h0 = blockIdx.y * 32;
    const int d0 = blockIdx.x * 32;
    const int t = threadIdx.x;
    const int r = t >> 3;
    const int c = (t & 7) * 4;

    float4 v = *(const float4*)(src + (size_t)(h0 + r) * DOUT + d0 + c);
    unsigned* p = (unsigned*)&tile[r][c];
    p[0] = pack2(v.x, v.y);
    p[1] = pack2(v.z, v.w);
    __syncthreads();
    unsigned lo = (unsigned)tile[c + 0][r] | ((unsigned)tile[c + 1][r] << 16);
    unsigned hi = (unsigned)tile[c + 2][r] | ((unsigned)tile[c + 3][r] << 16);
    uint2 o; o.x = lo; o.y = hi;
    *(uint2*)(dst + (size_t)(d0 + r) * HID + h0 + c) = o;
}

// ---------------------------------------------------------------------------
// Router: gate logits -> top-2 -> renormalized weights -> per-expert lists
// ---------------------------------------------------------------------------
__global__ __launch_bounds__(256)
void moe_router(const float* __restrict__ x, const float* __restrict__ gw,
                int* __restrict__ counts, int* __restrict__ tokidx,
                float* __restrict__ tokwgt)
{
    const int lane = threadIdx.x & 31;
    const int wave = threadIdx.x >> 5;
    const int tok  = blockIdx.x * 8 + wave;
    if (tok >= N_TOK) return;

    float p[NEXP];
#pragma unroll
    for (int e = 0; e < NEXP; ++e) p[e] = 0.f;

    const float* xr = x + (size_t)tok * HID;
    for (int i = lane; i < HID; i += 32) {
        float xv = xr[i];
#pragma unroll
        for (int e = 0; e < NEXP; ++e) p[e] += xv * gw[e * HID + i];
    }
#pragma unroll
    for (int e = 0; e < NEXP; ++e) {
        float v = p[e];
#pragma unroll
        for (int off = 16; off > 0; off >>= 1) v += __shfl_xor(v, off, 32);
        p[e] = v;
    }
    if (lane == 0) {
        int i0 = 0;
#pragma unroll
        for (int e = 1; e < NEXP; ++e) if (p[e] > p[i0]) i0 = e;
        int i1 = (i0 == 0) ? 1 : 0;
#pragma unroll
        for (int e = 0; e < NEXP; ++e) {
            if (e == i0 || e == i1) continue;
            if (p[e] > p[i1]) i1 = e;
        }
        float m  = p[i0];
        float e0 = __expf(p[i0] - m), e1 = __expf(p[i1] - m);
        float inv = 1.f / (e0 + e1);

        int pos0 = atomicAdd(&counts[i0], 1);
        tokidx[i0 * N_TOK + pos0] = tok;  tokwgt[i0 * N_TOK + pos0] = e0 * inv;
        int pos1 = atomicAdd(&counts[i1], 1);
        tokidx[i1 * N_TOK + pos1] = tok;  tokwgt[i1 * N_TOK + pos1] = e1 * inv;
    }
}

__global__ void moe_zero_counts(int* counts) {
    if (threadIdx.x < NEXP) counts[threadIdx.x] = 0;
}

// ---------------------------------------------------------------------------
// Tiled bf16 WMMA GEMM, async-to-LDS, 3 buffers, issue-ahead of 2 tiles.
//   ROUTED=false : out = xb @ WshT([D,H])               (plain stores)
//   ROUTED=true  : out[tok,:] += w_tok * gathered gemm  (f32 atomic adds)
// ---------------------------------------------------------------------------
template<bool ROUTED>
__global__ __launch_bounds__(256)
__attribute__((amdgpu_waves_per_eu(1)))
void moe_gemm(const unsigned short* __restrict__ xb, const unsigned short* __restrict__ Wt,
              float* __restrict__ out, const int* __restrict__ counts,
              const int* __restrict__ tokidx, const float* __restrict__ tokwgt)
{
    __shared__ __attribute__((aligned(16))) unsigned short As[3][BM][LDSP];
    __shared__ __attribute__((aligned(16))) unsigned short Bs[3][BN][LDSP];
    __shared__ int   tid_s[BM];
    __shared__ float twt_s[BM];

    const int e = ROUTED ? blockIdx.z : 0;
    const unsigned short* W = ROUTED ? (Wt + (size_t)e * HID * DOUT) : Wt;
    const int cnt = ROUTED ? counts[e] : N_TOK;
    const int m_base = blockIdx.y * BM;
    if (m_base >= cnt) return;
    const int n_base = blockIdx.x * BN;

    const int t = threadIdx.x;
    if (t < BM) {
        int row = m_base + t;
        int tok; float w;
        if (ROUTED) {
            if (row < cnt) { tok = tokidx[e * N_TOK + row]; w = tokwgt[e * N_TOK + row]; }
            else           { tok = -1; w = 0.f; }
        } else { tok = row; w = 1.f; }
        tid_s[t] = tok;  twt_s[t] = w;
    }
    __syncthreads();

    // Loader mapping: 2 threads per row/col, each covers 16 k elems (32B)
    const int lrow  = t >> 1;          // 0..127 : A row == B col index
    const int kbase = (t & 1) * 16;    // 0 or 16 within BK=32
    const int tokc  = max(tid_s[lrow], 0);       // clamp gather (bad rows skipped later)
    const unsigned aVoff0 = (unsigned)tokc * (HID * 2) + (unsigned)kbase * 2;
    const unsigned bVoff0 = (unsigned)(n_base + lrow) * (HID * 2) + (unsigned)kbase * 2;
    unsigned ldsA[3], ldsB[3];
#pragma unroll
    for (int b = 0; b < 3; ++b) {
        ldsA[b] = lds_off(&As[b][lrow][kbase]);
        ldsB[b] = lds_off(&Bs[b][lrow][kbase]);
    }

    const int lane = t & 31;
    const int wave = t >> 5;
    const int wm = (wave & 3) * 32;
    const int wn = (wave >> 2) * 64;
    const int half = lane >> 4;
    const int l16  = lane & 15;

    v8f acc[2][4];
#pragma unroll
    for (int i = 0; i < 2; ++i)
#pragma unroll
        for (int j = 0; j < 4; ++j) acc[i][j] = (v8f)0.f;

    // Issue tile kt into buffer `buf` (4 async instrs per wave)
    auto issue_tile = [&](int buf, int kt) {
        unsigned kB = (unsigned)(kt * BK * 2);
        async_copy_32B(ldsA[buf], aVoff0 + kB, xb);
        async_copy_32B(ldsB[buf], bVoff0 + kB, W);
    };
    // 8 WMMAs from buffer `buf`
    auto compute = [&](int buf) {
        v16bf afrag[2], bfrag[4];
#pragma unroll
        for (int i = 0; i < 2; ++i) {
            const unsigned short* base = &As[buf][wm + i * 16 + l16][half * 8];
            v8bf lo = *(const v8bf*)(base);        // K = half*8 + 0..7
            v8bf hi = *(const v8bf*)(base + 16);   // K = 16 + half*8 + 0..7
            afrag[i] = __builtin_shufflevector(lo, hi,
                0,1,2,3,4,5,6,7,8,9,10,11,12,13,14,15);
        }
#pragma unroll
        for (int j = 0; j < 4; ++j) {
            const unsigned short* base = &Bs[buf][wn + j * 16 + l16][half * 8];
            v8bf lo = *(const v8bf*)(base);        // K = half*8 + 0..7
            v8bf hi = *(const v8bf*)(base + 16);   // K = 16 + half*8 + 0..7
            bfrag[j] = __builtin_shufflevector(lo, hi,
                0,1,2,3,4,5,6,7,8,9,10,11,12,13,14,15);
        }
#pragma unroll
        for (int i = 0; i < 2; ++i)
#pragma unroll
            for (int j = 0; j < 4; ++j)
                acc[i][j] = __builtin_amdgcn_wmma_f32_16x16x32_bf16(
                    false, afrag[i], false, bfrag[j],
                    (short)0, acc[i][j], false, false);
    };

    constexpr int KT = HID / BK;   // 64 (64 % 3 == 1: main loop does 63, peel 1)
    issue_tile(0, 0);
    issue_tile(1, 1);
    // Steady state: in flight = tile kt (oldest 4) + tile kt+1 (newest 4).
    // wait_async4 -> tile kt resident for this wave; barrier -> for all waves,
    // and everyone is done reading the buffer tile kt+2 will overwrite.
#pragma unroll 1
    for (int kt = 0; kt < KT - 1; kt += 3) {
#pragma unroll
        for (int u = 0; u < 3; ++u) {
            const int k = kt + u;          // < KT-1 guaranteed for u<..? guarded below
            if (k >= KT - 1) break;
            wait_async4();
            __syncthreads();
            if (k + 2 < KT) issue_tile((k + 2) % 3, k + 2);
            compute(k % 3);
        }
    }
    // Peeled last tile: nothing newer in flight -> full wait.
    wait_async0();
    __syncthreads();
    compute((KT - 1) % 3);

    // Epilogue. C/D layout: lanes 0-15 -> M=v, N=lane ; lanes 16-31 -> M=8+v
#pragma unroll
    for (int i = 0; i < 2; ++i) {
#pragma unroll
        for (int v = 0; v < 8; ++v) {
            int rloc = wm + i * 16 + half * 8 + v;
            int tok  = tid_s[rloc];
            if (tok < 0) continue;
            float w = twt_s[rloc];
#pragma unroll
            for (int j = 0; j < 4; ++j) {
                float val = acc[i][j][v];
                float* dst = out + (size_t)tok * DOUT + n_base + wn + j * 16 + l16;
                if (ROUTED) atomicAdd(dst, w * val);
                else        *dst = val;
            }
        }
    }
}

// ---------------------------------------------------------------------------
extern "C" void kernel_launch(void* const* d_in, const int* in_sizes, int n_in,
                              void* d_out, int out_size, void* d_ws, size_t ws_size,
                              hipStream_t stream)
{
    const float* x           = (const float*)d_in[0];   // [N, H]
    const float* gate_weight = (const float*)d_in[1];   // [E, H]
    const float* W_routed    = (const float*)d_in[2];   // [E, H, D]
    const float* W_shared    = (const float*)d_in[3];   // [H, D]
    float* out = (float*)d_out;                         // [N, D]
    (void)in_sizes; (void)n_in; (void)out_size; (void)ws_size;

    // workspace layout (~110 MB)
    char* ws = (char*)d_ws;
    int*   counts = (int*)ws;                      ws += 256;
    int*   tokidx = (int*)ws;                      ws += sizeof(int)   * (size_t)NEXP * N_TOK;
    float* tokwgt = (float*)ws;                    ws += sizeof(float) * (size_t)NEXP * N_TOK;
    unsigned short* xb   = (unsigned short*)ws;    ws += 2ull * N_TOK * HID;
    unsigned short* wshT = (unsigned short*)ws;    ws += 2ull * HID * DOUT;
    unsigned short* wrtT = (unsigned short*)ws;    ws += 2ull * NEXP * HID * DOUT;

    moe_zero_counts<<<1, 32, 0, stream>>>(counts);
    convert_x_bf16<<<(N_TOK * HID) / (256 * 8), 256, 0, stream>>>(x, xb);
    convert_w_t_bf16<<<dim3(DOUT / 32, HID / 32, NEXP + 1), 256, 0, stream>>>(
        W_shared, W_routed, wshT, wrtT);
    moe_router<<<N_TOK / 8, 256, 0, stream>>>(x, gate_weight, counts, tokidx, tokwgt);

    dim3 gs(DOUT / BN, N_TOK / BM, 1);
    moe_gemm<false><<<gs, 256, 0, stream>>>(xb, wshT, out, counts, tokidx, tokwgt);

    dim3 ge(DOUT / BN, N_TOK / BM, NEXP);   // worst-case M grid; empty tiles exit early
    moe_gemm<true><<<ge, 256, 0, stream>>>(xb, wrtT, out, counts, tokidx, tokwgt);
}
